// GAT_model_23819888623725
// MI455X (gfx1250) — matrix operand order repaired and checked by reading, not compiled, not run on hardware
//
#include <hip/hip_runtime.h>
#include <hip/hip_bf16.h>

typedef __attribute__((ext_vector_type(16))) _Float16 v16h;
typedef __attribute__((ext_vector_type(8)))  _Float16 v8h;
typedef __attribute__((ext_vector_type(4)))  _Float16 v4h;
typedef __attribute__((ext_vector_type(8)))  float    v8f;

#define IN_EPS 1e-5f

__device__ __forceinline__ float eluf(float x) { return x > 0.f ? x : __expf(x) - 1.f; }

// Order-preserving float<->uint encode for atomicMax on floats
__device__ __forceinline__ unsigned encf(float f) {
    unsigned u = __float_as_uint(f);
    return (u & 0x80000000u) ? ~u : (u | 0x80000000u);
}
__device__ __forceinline__ float decf(unsigned u) {
    return __uint_as_float((u & 0x80000000u) ? (u & 0x7FFFFFFFu) : ~u);
}

// ---------------------------------------------------------------------------
// Per-column mean / mean-of-squares over N rows of an N x 64 matrix.
// ---------------------------------------------------------------------------
__global__ void colstats_k(const float* __restrict__ X, int n, int C,
                           float* __restrict__ mu, float* __restrict__ msq) {
    __shared__ float s1[256], s2[256];
    int c = blockIdx.x;
    float a = 0.f, b = 0.f;
    for (int i = threadIdx.x; i < n; i += blockDim.x) {
        float v = X[(size_t)i * C + c];
        a += v; b += v * v;
    }
    s1[threadIdx.x] = a; s2[threadIdx.x] = b;
    __syncthreads();
    for (int s = 128; s > 0; s >>= 1) {
        if ((int)threadIdx.x < s) { s1[threadIdx.x] += s1[threadIdx.x + s]; s2[threadIdx.x] += s2[threadIdx.x + s]; }
        __syncthreads();
    }
    if (threadIdx.x == 0) { mu[c] = s1[0] / n; msq[c] = s2[0] / n; }
}

__global__ void mkss_k(const float* __restrict__ mu, const float* __restrict__ msq,
                       const float* __restrict__ w, const float* __restrict__ b,
                       float* __restrict__ scale, float* __restrict__ shift) {
    int c = threadIdx.x;
    float m   = mu[c];
    float var = msq[c] - m * m;
    float r   = rsqrtf(var + IN_EPS);
    float sc  = r * w[c];
    scale[c] = sc;
    shift[c] = b[c] - m * sc;
}

__global__ void affelu_k(const float* __restrict__ X, const float* __restrict__ scale,
                         const float* __restrict__ shift, float* __restrict__ Y, int total) {
    int i = blockIdx.x * blockDim.x + threadIdx.x;
    if (i >= total) return;
    int c = i & 63;
    Y[i] = eluf(X[i] * scale[c] + shift[c]);
}

__global__ void resnorm_k(const float* __restrict__ h2, const float* __restrict__ g,
                          const float* __restrict__ scale, const float* __restrict__ shift,
                          float* __restrict__ h3, int total) {
    int i = blockIdx.x * blockDim.x + threadIdx.x;
    if (i >= total) return;
    int c = i & 63;
    h3[i] = eluf(h2[i] + g[i] * scale[c] + shift[c]);
}

__global__ void degdiv_k(float* __restrict__ agg, const float* __restrict__ deg, int total) {
    int i = blockIdx.x * blockDim.x + threadIdx.x;
    if (i >= total) return;
    agg[i] = agg[i] / deg[i >> 6];
}

__global__ void concat_k(const float* __restrict__ a, const float* __restrict__ b,
                         float* __restrict__ cat, int N) {
    int i = blockIdx.x * blockDim.x + threadIdx.x;
    if (i >= N * 128) return;
    int r = i >> 7, c = i & 127;
    cat[i] = (c < 64) ? a[r * 64 + c] : b[r * 64 + (c - 64)];
}

// final 16 -> 2 projection (too narrow for WMMA)
__global__ void dec_k(const float* __restrict__ d5, const float* __restrict__ W,
                      const float* __restrict__ bv, float* __restrict__ out, int N) {
    int n = blockIdx.x * blockDim.x + threadIdx.x;
    if (n >= N) return;
    float o0 = bv[0], o1 = bv[1];
#pragma unroll
    for (int k = 0; k < 16; k++) {
        float v = d5[n * 16 + k];
        o0 += v * W[k * 2 + 0];
        o1 += v * W[k * 2 + 1];
    }
    out[n * 2 + 0] = o0;
    out[n * 2 + 1] = o1;
}

// Transpose + f32->f16 convert a K x Nd weight into Wt[Nd][K]
__global__ void wtpose_k(const float* __restrict__ W, int K, int Nd,
                         _Float16* __restrict__ Wt) {
    int i = blockIdx.x * blockDim.x + threadIdx.x;
    if (i >= K * Nd) return;
    int k = i / Nd, n = i - k * Nd;
    Wt[(size_t)n * K + k] = (_Float16)W[i];
}

// ---------------------------------------------------------------------------
// Edge phase (GATv2). Etot = E real edges + N self loops (FILL edge attr).
// ---------------------------------------------------------------------------
__global__ void edge_logits_k(const int* __restrict__ ei, const float* __restrict__ ea,
                              const float* __restrict__ We, const float* __restrict__ xl,
                              const float* __restrict__ xr, const float* __restrict__ att,
                              float* __restrict__ logits, unsigned* __restrict__ lmax,
                              int E, int N) {
    int e = blockIdx.x * blockDim.x + threadIdx.x;
    int Et = E + N;
    if (e >= Et) return;
    int s, d;
    float eav[7];
    if (e < E) {
        s = ei[e]; d = ei[E + e];
#pragma unroll
        for (int j = 0; j < 7; j++) eav[j] = ea[(size_t)e * 7 + j];
    } else {
        s = d = e - E;
#pragma unroll
        for (int j = 0; j < 6; j++) eav[j] = 0.f;
        eav[6] = 1.f;
    }
    const float4* xls = (const float4*)(xl + (size_t)s * 64);
    const float4* xrd = (const float4*)(xr + (size_t)d * 64);
    const float4* We4 = (const float4*)We;
    float acc[4] = {0.f, 0.f, 0.f, 0.f};
#pragma unroll 4
    for (int q = 0; q < 16; q++) {
        float4 a = xls[q], b = xrd[q];
        float mv[4] = {a.x + b.x, a.y + b.y, a.z + b.z, a.w + b.w};
#pragma unroll
        for (int j = 0; j < 7; j++) {
            float4 w = We4[j * 16 + q];
            mv[0] += eav[j] * w.x; mv[1] += eav[j] * w.y;
            mv[2] += eav[j] * w.z; mv[3] += eav[j] * w.w;
        }
#pragma unroll
        for (int t = 0; t < 4; t++) {
            int c = q * 4 + t;
            float m = mv[t];
            m = (m > 0.f) ? m : 0.2f * m;            // leaky_relu(0.2)
            acc[c >> 4] += m * att[c];               // att is (4,16) row-major
        }
    }
#pragma unroll
    for (int h = 0; h < 4; h++) {
        logits[(size_t)e * 4 + h] = acc[h];
        atomicMax(&lmax[(size_t)d * 4 + h], encf(acc[h]));
    }
}

__global__ void edge_exp_k(const int* __restrict__ ei, float* __restrict__ logits,
                           const unsigned* __restrict__ lmax, float* __restrict__ denom,
                           float* __restrict__ deg, int E, int N) {
    int e = blockIdx.x * blockDim.x + threadIdx.x;
    int Et = E + N;
    if (e >= Et) return;
    int d = (e < E) ? ei[E + e] : (e - E);
#pragma unroll
    for (int h = 0; h < 4; h++) {
        float l  = logits[(size_t)e * 4 + h];
        float ex = __expf(l - decf(lmax[(size_t)d * 4 + h]));
        logits[(size_t)e * 4 + h] = ex;
        atomicAdd(&denom[(size_t)d * 4 + h], ex);
    }
    atomicAdd(&deg[d], 1.f);
}

__global__ void edge_agg_k(const int* __restrict__ ei, const float* __restrict__ exv,
                           const float* __restrict__ denom, const float* __restrict__ xl,
                           float* __restrict__ agg, int E, int N) {
    int e = blockIdx.x * blockDim.x + threadIdx.x;
    int Et = E + N;
    if (e >= Et) return;
    int s, d;
    if (e < E) { s = ei[e]; d = ei[E + e]; } else { s = d = e - E; }
    float al[4];
#pragma unroll
    for (int h = 0; h < 4; h++)
        al[h] = exv[(size_t)e * 4 + h] / denom[(size_t)d * 4 + h];
    const float4* xls = (const float4*)(xl + (size_t)s * 64);
    float* ag = agg + (size_t)d * 64;
#pragma unroll 4
    for (int q = 0; q < 16; q++) {
        float4 v = xls[q];
        float a  = al[q >> 2];
        atomicAdd(&ag[q * 4 + 0], a * v.x);
        atomicAdd(&ag[q * 4 + 1], a * v.y);
        atomicAdd(&ag[q * 4 + 2], a * v.z);
        atomicAdd(&ag[q * 4 + 3], a * v.w);
    }
}

// ---------------------------------------------------------------------------
// WMMA GEMM: Y[N x ND] = act_out( A @ W (+ bias) ),  A = act_in(X*scale+shift)
// X: N x KD f32.  Wt: ND x KD f16 (pre-transposed weights).
// One block per 16-row tile; wave w handles column tile w (ND/16 waves).
// A tile staged once in LDS as f16; output tile bounced through LDS for
// coalesced b128 stores.  AMODE: 0 raw, 1 affine, 2 affine+ELU. OACT: 0/1=ELU.
// ---------------------------------------------------------------------------
template <int KD, int ND, int AMODE, int OACT, bool BIAS>
__global__ __launch_bounds__(ND * 2) void gemm_wmma(
    const float* __restrict__ X, int nrows,
    const float* __restrict__ scale, const float* __restrict__ shift,
    const _Float16* __restrict__ Wt, const float* __restrict__ bias,
    float* __restrict__ Y) {
    constexpr int LDA    = KD + 16;  // halves; rows stay 16B aligned
    constexpr int ABYTES = 16 * LDA * 2;
    constexpr int OBYTES = 16 * ND * 4;
    constexpr int SBYTES = ABYTES > OBYTES ? ABYTES : OBYTES;
    __shared__ __align__(16) char smem[SBYTES];
    _Float16* As = (_Float16*)smem;
    float*    Os = (float*)smem;

    const int r0 = blockIdx.x * 16;
    const bool full = (r0 + 16 <= nrows);

    // ---- stage A tile (float4 loads, fused transform, f16 LDS stores) ----
    for (int idx = threadIdx.x; idx < 16 * (KD / 4); idx += blockDim.x) {
        int r = idx / (KD / 4), q = idx - r * (KD / 4);
        int k = q * 4;
        int gr = r0 + r;
        float4 v = make_float4(0.f, 0.f, 0.f, 0.f);
        if (gr < nrows) v = *(const float4*)&X[(size_t)gr * KD + k];
        float vv[4] = {v.x, v.y, v.z, v.w};
        if (AMODE >= 1) {
            float4 s4 = *(const float4*)&scale[k];
            float4 h4 = *(const float4*)&shift[k];
            vv[0] = vv[0] * s4.x + h4.x; vv[1] = vv[1] * s4.y + h4.y;
            vv[2] = vv[2] * s4.z + h4.z; vv[3] = vv[3] * s4.w + h4.w;
        }
        if (AMODE == 2) {
#pragma unroll
            for (int t = 0; t < 4; t++) vv[t] = eluf(vv[t]);
        }
        v4h hv;
#pragma unroll
        for (int t = 0; t < 4; t++) hv[t] = (_Float16)vv[t];
        *(v4h*)&As[r * LDA + k] = hv;
    }
    __syncthreads();

    const int lane = threadIdx.x & 31;
    const int wave = threadIdx.x >> 5;
    const int n0   = wave * 16;
    const int rr   = lane & 15;        // A row / B col within tile
    const int hi   = lane >> 4;        // half-wave select
    const int kbA  = hi * 8;           // A: K {0..7,16..23} vs {8..15,24..31}
    const int kbB  = hi * 8;           // B: K {0..7,16..23} vs {8..15,24..31}

    const _Float16* wrow = Wt + (size_t)(n0 + rr) * KD;

    v8f c = {};
    for (int kk = 0; kk < KD; kk += 32) {
        v8h alo = *(const v8h*)&As[rr * LDA + kk + kbA];
        v8h ahi = *(const v8h*)&As[rr * LDA + kk + kbA + 16];
        v16h a;
#pragma unroll
        for (int i = 0; i < 8; i++) { a[i] = alo[i]; a[8 + i] = ahi[i]; }
        v8h blo = *(const v8h*)&wrow[kk + kbB];
        v8h bhi = *(const v8h*)&wrow[kk + kbB + 16];
        v16h b;
#pragma unroll
        for (int i = 0; i < 8; i++) { b[i] = blo[i]; b[8 + i] = bhi[i]; }
        c = __builtin_amdgcn_wmma_f32_16x16x32_f16(false, a, false, b,
                                                   (short)0, c, false, false);
    }

    // ---- epilogue through LDS for coalesced stores ----
    __syncthreads();   // all waves done reading As before overwriting
    float bn = BIAS ? bias[n0 + rr] : 0.f;
#pragma unroll
    for (int v = 0; v < 8; v++) {
        int m = v + hi * 8;            // C/D: VGPR v holds row m (lanes>=16: +8)
        float o = c[v] + bn;
        if (OACT == 1) o = eluf(o);
        Os[m * ND + n0 + rr] = o;
    }
    __syncthreads();

    if (full) {
        for (int idx = threadIdx.x; idx < 16 * (ND / 4); idx += blockDim.x) {
            int r = idx / (ND / 4), q = idx - r * (ND / 4);
            *(float4*)&Y[(size_t)(r0 + r) * ND + q * 4] = *(const float4*)&Os[r * ND + q * 4];
        }
    } else {
        for (int idx = threadIdx.x; idx < 16 * ND; idx += blockDim.x) {
            int r = idx / ND, cc = idx - r * ND;
            if (r0 + r < nrows) Y[(size_t)(r0 + r) * ND + cc] = Os[idx];
        }
    }
}

// ---------------------------------------------------------------------------
extern "C" void kernel_launch(void* const* d_in, const int* in_sizes, int n_in,
                              void* d_out, int out_size, void* d_ws, size_t ws_size,
                              hipStream_t stream) {
    const float* x    = (const float*)d_in[0];
    const int*   ei   = (const int*)d_in[1];
    const float* ea   = (const float*)d_in[2];
    const float* pre_w = (const float*)d_in[3];
    const float* pre_b = (const float*)d_in[4];
    const float* eW1  = (const float*)d_in[5];
    const float* en1w = (const float*)d_in[6];
    const float* en1b = (const float*)d_in[7];
    const float* eW2  = (const float*)d_in[8];
    const float* en2w = (const float*)d_in[9];
    const float* en2b = (const float*)d_in[10];
    const float* Wl   = (const float*)d_in[11];
    const float* Wr   = (const float*)d_in[12];
    const float* We   = (const float*)d_in[13];
    const float* att  = (const float*)d_in[14];
    const float* gn_w = (const float*)d_in[15];
    const float* gn_b = (const float*)d_in[16];
    const float* dW1 = (const float*)d_in[17]; const float* db1 = (const float*)d_in[18];
    const float* dW2 = (const float*)d_in[19]; const float* db2 = (const float*)d_in[20];
    const float* dW3 = (const float*)d_in[21]; const float* db3 = (const float*)d_in[22];
    const float* dW4 = (const float*)d_in[23]; const float* db4 = (const float*)d_in[24];
    const float* dW5 = (const float*)d_in[25]; const float* db5 = (const float*)d_in[26];
    const float* dW6 = (const float*)d_in[27]; const float* db6 = (const float*)d_in[28];
    const float* rW1 = (const float*)d_in[29]; const float* rb1 = (const float*)d_in[30];
    const float* rW2 = (const float*)d_in[31]; const float* rb2 = (const float*)d_in[32];
    const float* rW3 = (const float*)d_in[33]; const float* rb3 = (const float*)d_in[34];

    const int N  = in_sizes[0] / 64;
    const int E  = in_sizes[1] / 2;
    const int Et = E + N;
    const size_t N64 = (size_t)N * 64;

    // ---- workspace layout (float units) ----
    float* ws = (float*)d_ws;
    size_t off = 0;
    auto alloc = [&](size_t n) { float* p = ws + off; off += n; return p; };
    float*    tA    = alloc(N64);               // t1, later h3
    float*    tB    = alloc(N64);               // t2, later d3
    float*    h2    = alloc(N64);               // jk
    float*    xl    = alloc(N64);               // later d2 (with xr: N x 128)
    float*    xr    = alloc(N64);
    float*    agg   = alloc(N64);               // gat_out, later d4/d5
    float*    exb   = alloc((size_t)Et * 4);    // logits -> exp; later cat
    unsigned* lmax  = (unsigned*)alloc((size_t)N * 4);
    float*    denom = alloc((size_t)N * 4);
    float*    deg   = alloc(N);
    float*    d1    = alloc((size_t)N * 256);   // later r1/r2
    float*    mu    = alloc(64);
    float*    msq   = alloc(64);
    float*    sc    = alloc(64);
    float*    sh    = alloc(64);
    // f16 pre-transposed weights
    _Float16* hwts = (_Float16*)alloc(60000);   // 120000 halves available
    size_t ho = 0;
    auto halloc = [&](size_t n) { _Float16* p = hwts + ho; ho += n; return p; };
    _Float16* eW1t = halloc(4096);
    _Float16* eW2t = halloc(4096);
    _Float16* Wlt  = halloc(4096);
    _Float16* Wrt  = halloc(4096);
    _Float16* dW1t = halloc(32768);
    _Float16* dW2t = halloc(32768);
    _Float16* dW3t = halloc(8192);
    _Float16* dW4t = halloc(2048);
    _Float16* dW5t = halloc(512);
    _Float16* rW1t = halloc(4096);
    _Float16* rW2t = halloc(4096);
    _Float16* rW3t = halloc(4096);

    float* cat = exb;                  // N x 128 <= Et*4
    float* d2  = xl;                   // N x 128 (xl+xr contiguous)
    float* d3  = tB;                   // N x 64
    float* d4  = agg;                  // N x 32
    float* d5  = agg + (size_t)N * 32; // N x 16
    float* h3  = tA;
    float* r1  = d1;
    float* r2  = d1 + N64;
    float* dec = (float*)d_out;
    float* rec = (float*)d_out + (size_t)N * 2;

    const int RT  = (N + 15) / 16;
    const int tot = N * 64;
    const int eb  = (tot + 255) / 256;
    const int ebk = (Et + 255) / 256;

    // ---- pre-transpose all weights to f16 (tiny; independent) ----
    wtpose_k<<<16, 256, 0, stream>>>(eW1, 64, 64, eW1t);
    wtpose_k<<<16, 256, 0, stream>>>(eW2, 64, 64, eW2t);
    wtpose_k<<<16, 256, 0, stream>>>(Wl, 64, 64, Wlt);
    wtpose_k<<<16, 256, 0, stream>>>(Wr, 64, 64, Wrt);
    wtpose_k<<<128, 256, 0, stream>>>(dW1, 128, 256, dW1t);
    wtpose_k<<<128, 256, 0, stream>>>(dW2, 256, 128, dW2t);
    wtpose_k<<<32, 256, 0, stream>>>(dW3, 128, 64, dW3t);
    wtpose_k<<<8, 256, 0, stream>>>(dW4, 64, 32, dW4t);
    wtpose_k<<<2, 256, 0, stream>>>(dW5, 32, 16, dW5t);
    wtpose_k<<<16, 256, 0, stream>>>(rW1, 64, 64, rW1t);
    wtpose_k<<<16, 256, 0, stream>>>(rW2, 64, 64, rW2t);
    wtpose_k<<<16, 256, 0, stream>>>(rW3, 64, 64, rW3t);

    // ---- encoder: inorm -> gemm -> inorm+elu -> gemm -> inorm+elu ----
    colstats_k<<<64, 256, 0, stream>>>(x, N, 64, mu, msq);
    mkss_k<<<1, 64, 0, stream>>>(mu, msq, pre_w, pre_b, sc, sh);
    gemm_wmma<64, 64, 1, 0, false><<<RT, 128, 0, stream>>>(x, N, sc, sh, eW1t, nullptr, tA);

    colstats_k<<<64, 256, 0, stream>>>(tA, N, 64, mu, msq);
    mkss_k<<<1, 64, 0, stream>>>(mu, msq, en1w, en1b, sc, sh);
    gemm_wmma<64, 64, 2, 0, false><<<RT, 128, 0, stream>>>(tA, N, sc, sh, eW2t, nullptr, tB);

    colstats_k<<<64, 256, 0, stream>>>(tB, N, 64, mu, msq);
    mkss_k<<<1, 64, 0, stream>>>(mu, msq, en2w, en2b, sc, sh);
    affelu_k<<<eb, 256, 0, stream>>>(tB, sc, sh, h2, tot);

    // ---- GATv2 ----
    gemm_wmma<64, 64, 0, 0, false><<<RT, 128, 0, stream>>>(h2, N, nullptr, nullptr, Wlt, nullptr, xl);
    gemm_wmma<64, 64, 0, 0, false><<<RT, 128, 0, stream>>>(h2, N, nullptr, nullptr, Wrt, nullptr, xr);

    (void)hipMemsetAsync(lmax, 0, (size_t)N * 4 * sizeof(unsigned), stream);
    (void)hipMemsetAsync(denom, 0, (size_t)N * 4 * sizeof(float), stream);
    (void)hipMemsetAsync(deg, 0, (size_t)N * sizeof(float), stream);
    (void)hipMemsetAsync(agg, 0, N64 * sizeof(float), stream);

    edge_logits_k<<<ebk, 256, 0, stream>>>(ei, ea, We, xl, xr, att, exb, lmax, E, N);
    edge_exp_k<<<ebk, 256, 0, stream>>>(ei, exb, lmax, denom, deg, E, N);
    edge_agg_k<<<ebk, 256, 0, stream>>>(ei, exb, denom, xl, agg, E, N);
    degdiv_k<<<eb, 256, 0, stream>>>(agg, deg, tot);

    colstats_k<<<64, 256, 0, stream>>>(agg, N, 64, mu, msq);
    mkss_k<<<1, 64, 0, stream>>>(mu, msq, gn_w, gn_b, sc, sh);
    resnorm_k<<<eb, 256, 0, stream>>>(h2, agg, sc, sh, h3, tot);

    // ---- decoder ----
    concat_k<<<(N * 128 + 255) / 256, 256, 0, stream>>>(h2, h3, cat, N);
    gemm_wmma<128, 256, 0, 1, true><<<RT, 512, 0, stream>>>(cat, N, nullptr, nullptr, dW1t, db1, d1);
    gemm_wmma<256, 128, 0, 1, true><<<RT, 256, 0, stream>>>(d1, N, nullptr, nullptr, dW2t, db2, d2);
    gemm_wmma<128, 64, 0, 1, true><<<RT, 128, 0, stream>>>(d2, N, nullptr, nullptr, dW3t, db3, d3);
    gemm_wmma<64, 32, 0, 1, true><<<RT, 64, 0, stream>>>(d3, N, nullptr, nullptr, dW4t, db4, d4);
    gemm_wmma<32, 16, 0, 1, true><<<RT, 32, 0, stream>>>(d4, N, nullptr, nullptr, dW5t, db5, d5);
    dec_k<<<(N + 255) / 256, 256, 0, stream>>>(d5, dW6, db6, dec, N);

    // ---- recon head (writes directly into d_out tail) ----
    gemm_wmma<64, 64, 0, 1, true><<<RT, 128, 0, stream>>>(h3, N, nullptr, nullptr, rW1t, rb1, r1);
    gemm_wmma<64, 64, 0, 1, true><<<RT, 128, 0, stream>>>(r1, N, nullptr, nullptr, rW2t, rb2, r2);
    gemm_wmma<64, 64, 0, 0, true><<<RT, 128, 0, stream>>>(r2, N, nullptr, nullptr, rW3t, rb3, rec);
}
